// DrugGene_network_21921513079317
// MI455X (gfx1250) — compile-verified, matrix operand order checked by reading, not run on hardware
//
#include <hip/hip_runtime.h>
#include <hip/hip_bf16.h>

// ---------------------------------------------------------------------------
// DrugGene network for MI455X (gfx1250).
// Dominant cost: gene einsum = GEMM (4096x3008)x(3008x5000) = 123 GFLOP,
// ~50x compute-bound vs 23.3 TB/s HBM. Run it on v_wmma_f32_16x16x32_bf16
// using a bf16 hi/lo split (3 WMMAs per K=32) for fp32-grade accuracy.
// Everything else is tiny and runs as VALU kernels with BN fused into
// consumers (two-pass batch norm: stats kernel + on-the-fly apply).
// ---------------------------------------------------------------------------

typedef __bf16 bf16;
typedef __attribute__((ext_vector_type(16))) __bf16 v16bf;
typedef __attribute__((ext_vector_type(8)))  __bf16 v8bf;
typedef __attribute__((ext_vector_type(8)))  float  v8f;

#define WMMA_BF16(a, b, c) \
  __builtin_amdgcn_wmma_f32_16x16x32_bf16(false, (a), false, (b), (short)0, (c), false, false)

// Problem constants
#define BATCH      4096
#define NGENE      3008
#define NDRUG      2048
#define XCOLS      (NGENE + NDRUG)   // 5056
#define NGCOLS     5000              // 625 terms * 8 genes-per-term
#define NGPAD      5056              // 79 * 64
#define NDOUT0     100               // drug layer 1 width
#define NDPAD0     128               // padded to 2 * 64
#define EPSV       1e-5f

// ---------------------------------------------------------------------------
// bf16 hi/lo split helpers
// ---------------------------------------------------------------------------
__device__ __forceinline__ void split_bf16(float v, bf16& hi, bf16& lo) {
  bf16 h = (bf16)v;
  hi = h;
  lo = (bf16)(v - (float)h);
}

// Per-lane WMMA fragment load from K-major storage.
// 16-bit A (and mirrored B) fragment layout (ISA 7.12.2):
//   lane L: row/col = L & 15; chunk base c = 8*(L>>4);
//   elements 0..7  = mat[row][k0 + c .. c+7]
//   elements 8..15 = mat[row][k0 + c+16 .. c+23]
__device__ __forceinline__ v16bf load_frag(const bf16* __restrict__ base,
                                           int row, int K, int k0, int lane) {
  const bf16* p = base + (size_t)row * (size_t)K + (size_t)(k0 + ((lane >> 4) << 3));
  union { v16bf v; v8bf h[2]; } u;
  u.h[0] = *(const v8bf*)(p);
  u.h[1] = *(const v8bf*)(p + 16);
  return u.v;
}

// ---------------------------------------------------------------------------
// Split kernels (fp32 -> bf16 hi/lo, K-major, zero-padded rows)
// ---------------------------------------------------------------------------
__global__ void split_x_kernel(const float* __restrict__ x,
                               bf16* __restrict__ Ahi, bf16* __restrict__ Alo,
                               bf16* __restrict__ Dhi, bf16* __restrict__ Dlo) {
  int idx = blockIdx.x * blockDim.x + threadIdx.x;
  if (idx >= BATCH * XCOLS) return;
  int b = idx / XCOLS, n = idx % XCOLS;
  float v = x[idx];
  bf16 h, l;
  split_bf16(v, h, l);
  if (n < NGENE) {
    size_t o = (size_t)b * NGENE + n;
    Ahi[o] = h; Alo[o] = l;
  } else {
    size_t o = (size_t)b * NDRUG + (n - NGENE);
    Dhi[o] = h; Dlo[o] = l;
  }
}

__global__ void split_w_kernel(const float* __restrict__ W,
                               bf16* __restrict__ Whi, bf16* __restrict__ Wlo,
                               int rows_valid, int rows_pad, int K) {
  int idx = blockIdx.x * blockDim.x + threadIdx.x;
  if (idx >= rows_pad * K) return;
  int r = idx / K, k = idx % K;
  bf16 h = (bf16)0.0f, l = (bf16)0.0f;
  if (r < rows_valid) split_bf16(W[(size_t)r * K + k], h, l);
  Whi[idx] = h; Wlo[idx] = l;
}

// ---------------------------------------------------------------------------
// bf16x3 WMMA GEMM:  C[m][n] = sum_k A[m][k] * B[n][k]  (+ bias, opt tanh)
// A: M x K (K-major), B: Npad x K (K-major, zero-padded rows).
// Block = 8 waves, each wave computes a 16(M) x 64(N) tile (4 accumulators,
// A fragment reused across the 4 N sub-tiles -> 12 WMMAs per K=32 step).
// ---------------------------------------------------------------------------
__launch_bounds__(256)
__global__ void gemm_bf16x3_kernel(const bf16* __restrict__ Ahi, const bf16* __restrict__ Alo,
                                   const bf16* __restrict__ Bhi, const bf16* __restrict__ Blo,
                                   float* __restrict__ C, const float* __restrict__ bias,
                                   int K, int Nvalid, int do_tanh) {
  const int lane = threadIdx.x & 31;
  const int wave = threadIdx.x >> 5;
  const int mbase = blockIdx.y * 128 + wave * 16;
  const int nbase = blockIdx.x * 64;
  const int arow  = mbase + (lane & 15);
  const int brow  = nbase + (lane & 15);

  v8f acc[4] = {};

  for (int k0 = 0; k0 < K; k0 += 32) {
    // prefetch next K tile of A (gfx1250 global_prefetch_b8 path)
    if (k0 + 32 < K)
      __builtin_prefetch(Ahi + (size_t)arow * (size_t)K + (size_t)(k0 + 32), 0, 1);

    v16bf ah = load_frag(Ahi, arow, K, k0, lane);
    v16bf al = load_frag(Alo, arow, K, k0, lane);
#pragma unroll
    for (int s = 0; s < 4; ++s) {
      v16bf bh = load_frag(Bhi, brow + s * 16, K, k0, lane);
      v16bf bl = load_frag(Blo, brow + s * 16, K, k0, lane);
      acc[s] = WMMA_BF16(ah, bh, acc[s]);  // hi*hi
      acc[s] = WMMA_BF16(ah, bl, acc[s]);  // hi*lo
      acc[s] = WMMA_BF16(al, bh, acc[s]);  // lo*hi
    }
  }

  // D layout: VGPR r -> lanes 0-15: (M=r, N=lane); lanes 16-31: (M=r+8, N=lane-16)
  const int mtop = mbase + ((lane >> 4) << 3);
#pragma unroll
  for (int s = 0; s < 4; ++s) {
    int n = nbase + s * 16 + (lane & 15);
    if (n < Nvalid) {
      float bv = bias ? bias[n] : 0.0f;
#pragma unroll
      for (int r = 0; r < 8; ++r) {
        float v = acc[s][r] + bv;
        if (do_tanh) v = tanhf(v);
        C[(size_t)(mtop + r) * (size_t)Nvalid + (size_t)n] = v;
      }
    }
  }
}

// ---------------------------------------------------------------------------
// Batch-norm statistics: one block per column; mean and rsqrt(var+eps).
// ---------------------------------------------------------------------------
__global__ void stats_kernel(const float* __restrict__ z, float* __restrict__ mean,
                             float* __restrict__ inv, int C) {
  __shared__ float s1[256];
  __shared__ float s2[256];
  const int c = blockIdx.x;
  float a = 0.0f, q = 0.0f;
  for (int i = threadIdx.x; i < BATCH; i += 256) {
    float v = z[(size_t)i * (size_t)C + (size_t)c];
    a += v; q += v * v;
  }
  s1[threadIdx.x] = a; s2[threadIdx.x] = q;
  __syncthreads();
  for (int off = 128; off > 0; off >>= 1) {
    if (threadIdx.x < off) {
      s1[threadIdx.x] += s1[threadIdx.x + off];
      s2[threadIdx.x] += s2[threadIdx.x + off];
    }
    __syncthreads();
  }
  if (threadIdx.x == 0) {
    float m = s1[0] * (1.0f / BATCH);
    float v = s2[0] * (1.0f / BATCH) - m * m;
    mean[c] = m;
    inv[c]  = rsqrtf(v + EPSV);
  }
}

__device__ __forceinline__ float bn_apply(float z, float m, float iv, float g, float be) {
  return g * (z - m) * iv + be;
}

// ---------------------------------------------------------------------------
// Tree level 0: z0[b, t*6+h] = tanh( sum_g h0[b, t*8+g] * W[(t*6+h)*8+g] + b )
// ---------------------------------------------------------------------------
__global__ void term0_kernel(const float* __restrict__ h0, const float* __restrict__ W,
                             const float* __restrict__ bvec, float* __restrict__ z) {
  int idx = blockIdx.x * blockDim.x + threadIdx.x;
  const int C = 625 * 6;
  if (idx >= BATCH * C) return;
  int b = idx / C, r = idx % C;
  int t = r / 6;
  const float* hp = h0 + (size_t)b * NGCOLS + (size_t)t * 8;
  const float* wp = W + (size_t)r * 8;
  float a = bvec[r];
#pragma unroll
  for (int g = 0; g < 8; ++g) a += hp[g] * wp[g];
  z[(size_t)b * C + r] = tanhf(a);
}

// Tree levels 1..4: children of t' are columns [t'*30, t'*30+30) of prev level.
__global__ void term_next_kernel(const float* __restrict__ zprev,
                                 const float* __restrict__ mean, const float* __restrict__ inv,
                                 const float* __restrict__ gamma, const float* __restrict__ beta,
                                 const float* __restrict__ W, const float* __restrict__ bvec,
                                 float* __restrict__ z, int nterms, int Cprev) {
  const int C = nterms * 6;
  int idx = blockIdx.x * blockDim.x + threadIdx.x;
  if (idx >= BATCH * C) return;
  int b = idx / C, r = idx % C;
  int tp = r / 6;
  const float* wp = W + (size_t)r * 30;
  const int col0 = tp * 30;
  float a = bvec[r];
#pragma unroll
  for (int q = 0; q < 30; ++q) {
    int col = col0 + q;
    float v = bn_apply(zprev[(size_t)b * Cprev + col], mean[col], inv[col], gamma[col], beta[col]);
    a += v * wp[q];
  }
  z[(size_t)b * C + r] = tanhf(a);
}

// ---------------------------------------------------------------------------
// Drug FC layer (BN-applied input): z[b,o] = tanh( BN(zprev[b,:]) . W[o,:] + b[o] )
// ---------------------------------------------------------------------------
__global__ void fc_bn_tanh_kernel(const float* __restrict__ zprev,
                                  const float* __restrict__ mean, const float* __restrict__ inv,
                                  const float* __restrict__ gamma, const float* __restrict__ beta,
                                  const float* __restrict__ W, const float* __restrict__ bvec,
                                  float* __restrict__ z, int Cin, int Cout) {
  int idx = blockIdx.x * blockDim.x + threadIdx.x;
  if (idx >= BATCH * Cout) return;
  int b = idx / Cout, o = idx % Cout;
  const float* wp = W + (size_t)o * Cin;
  float a = bvec[o];
  for (int k = 0; k < Cin; ++k) {
    float v = bn_apply(zprev[(size_t)b * Cin + k], mean[k], inv[k], gamma[k], beta[k]);
    a += v * wp[k];
  }
  z[(size_t)b * Cout + o] = tanhf(a);
}

// Aux head: out[b] = tanh( BN(zd[b,:]) . W1 + b1 ) * W2 + b2
__global__ void aux_kernel(const float* __restrict__ zd,
                           const float* __restrict__ mean, const float* __restrict__ inv,
                           const float* __restrict__ gamma, const float* __restrict__ beta,
                           const float* __restrict__ W1, const float* __restrict__ b1,
                           const float* __restrict__ W2, const float* __restrict__ b2,
                           float* __restrict__ out, int Cin) {
  int b = blockIdx.x * blockDim.x + threadIdx.x;
  if (b >= BATCH) return;
  float a = b1[0];
  for (int k = 0; k < Cin; ++k) {
    float v = bn_apply(zd[(size_t)b * Cin + k], mean[k], inv[k], gamma[k], beta[k]);
    a += v * W1[k];
  }
  out[b] = tanhf(a) * W2[0] + b2[0];
}

// Final layer pre-BN: zf[b,h] = tanh( [BN(z4) ; BN(zd2)] . finW[h,:] + finb[h] )
__global__ void final_z_kernel(const float* __restrict__ z4,
                               const float* __restrict__ m4, const float* __restrict__ i4,
                               const float* __restrict__ g4, const float* __restrict__ be4,
                               const float* __restrict__ zd2,
                               const float* __restrict__ md, const float* __restrict__ id,
                               const float* __restrict__ gd, const float* __restrict__ bed,
                               const float* __restrict__ finW, const float* __restrict__ finb,
                               float* __restrict__ zf) {
  int idx = blockIdx.x * blockDim.x + threadIdx.x;
  if (idx >= BATCH * 6) return;
  int b = idx / 6, h = idx % 6;
  const float* wp = finW + (size_t)h * 12;
  float a = finb[h];
#pragma unroll
  for (int j = 0; j < 6; ++j)
    a += bn_apply(z4[(size_t)b * 6 + j], m4[j], i4[j], g4[j], be4[j]) * wp[j];
#pragma unroll
  for (int j = 0; j < 6; ++j)
    a += bn_apply(zd2[(size_t)b * 6 + j], md[j], id[j], gd[j], bed[j]) * wp[6 + j];
  zf[(size_t)b * 6 + h] = tanhf(a);
}

// ---------------------------------------------------------------------------
// Host-side orchestration
// ---------------------------------------------------------------------------
extern "C" void kernel_launch(void* const* d_in, const int* in_sizes, int n_in,
                              void* d_out, int out_size, void* d_ws, size_t ws_size,
                              hipStream_t stream) {
  (void)in_sizes; (void)n_in; (void)out_size; (void)ws_size;

  // ---- input pointers (setup_inputs dict order, lists flattened in order) ----
  const float* x    = (const float*)d_in[0];
  const float* dgW  = (const float*)d_in[1];   // (625, 8, 3008)
  const float* dgb  = (const float*)d_in[2];   // (625, 8) -> 5000
  const float* tW[5]  = {(const float*)d_in[3],  (const float*)d_in[4],  (const float*)d_in[5],
                         (const float*)d_in[6],  (const float*)d_in[7]};
  const float* tb[5]  = {(const float*)d_in[8],  (const float*)d_in[9],  (const float*)d_in[10],
                         (const float*)d_in[11], (const float*)d_in[12]};
  const float* tg[5]  = {(const float*)d_in[13], (const float*)d_in[14], (const float*)d_in[15],
                         (const float*)d_in[16], (const float*)d_in[17]};
  const float* tbe[5] = {(const float*)d_in[18], (const float*)d_in[19], (const float*)d_in[20],
                         (const float*)d_in[21], (const float*)d_in[22]};
  const float* dW[3]  = {(const float*)d_in[23], (const float*)d_in[24], (const float*)d_in[25]};
  const float* db[3]  = {(const float*)d_in[26], (const float*)d_in[27], (const float*)d_in[28]};
  const float* dg[3]  = {(const float*)d_in[29], (const float*)d_in[30], (const float*)d_in[31]};
  const float* dbe[3] = {(const float*)d_in[32], (const float*)d_in[33], (const float*)d_in[34]};
  const float* aW1[3] = {(const float*)d_in[35], (const float*)d_in[36], (const float*)d_in[37]};
  const float* ab1[3] = {(const float*)d_in[38], (const float*)d_in[39], (const float*)d_in[40]};
  const float* aW2[3] = {(const float*)d_in[41], (const float*)d_in[42], (const float*)d_in[43]};
  const float* ab2[3] = {(const float*)d_in[44], (const float*)d_in[45], (const float*)d_in[46]};
  const float* finW    = (const float*)d_in[47];
  const float* finb    = (const float*)d_in[48];
  const float* fing    = (const float*)d_in[49];
  const float* finbeta = (const float*)d_in[50];
  const float* finauxW = (const float*)d_in[51];
  const float* finauxb = (const float*)d_in[52];
  const float* foutW   = (const float*)d_in[53];
  const float* foutb   = (const float*)d_in[54];

  float* out = (float*)d_out;  // [aux_final | aux_drug0 | aux_drug1 | aux_drug2], 4096 each

  // ---- workspace carve-out (256B aligned) ----
  size_t off = 0;
  auto alloc = [&](size_t bytes) -> void* {
    void* p = (char*)d_ws + off;
    off += (bytes + 255) & ~(size_t)255;
    return p;
  };
  bf16* Ahi = (bf16*)alloc((size_t)BATCH * NGENE * 2);
  bf16* Alo = (bf16*)alloc((size_t)BATCH * NGENE * 2);
  bf16* Dhi = (bf16*)alloc((size_t)BATCH * NDRUG * 2);
  bf16* Dlo = (bf16*)alloc((size_t)BATCH * NDRUG * 2);
  bf16* Whi = (bf16*)alloc((size_t)NGPAD * NGENE * 2);
  bf16* Wlo = (bf16*)alloc((size_t)NGPAD * NGENE * 2);
  bf16* WDhi = (bf16*)alloc((size_t)NDPAD0 * NDRUG * 2);
  bf16* WDlo = (bf16*)alloc((size_t)NDPAD0 * NDRUG * 2);
  float* h0 = (float*)alloc((size_t)BATCH * NGCOLS * 4);

  const int Cterm[5] = {3750, 750, 150, 30, 6};
  float* zt[5]; float* mt[5]; float* it[5];
  for (int i = 0; i < 5; ++i) {
    zt[i] = (float*)alloc((size_t)BATCH * Cterm[i] * 4);
    mt[i] = (float*)alloc((size_t)Cterm[i] * 4);
    it[i] = (float*)alloc((size_t)Cterm[i] * 4);
  }
  const int Cdrug[3] = {100, 50, 6};
  float* zd[3]; float* md[3]; float* id[3];
  for (int i = 0; i < 3; ++i) {
    zd[i] = (float*)alloc((size_t)BATCH * Cdrug[i] * 4);
    md[i] = (float*)alloc((size_t)Cdrug[i] * 4);
    id[i] = (float*)alloc((size_t)Cdrug[i] * 4);
  }
  float* zf = (float*)alloc((size_t)BATCH * 6 * 4);
  float* mf = (float*)alloc(6 * 4);
  float* iff = (float*)alloc(6 * 4);

  const int T = 256;
  auto blocks = [](long long n) { return (unsigned)((n + 255) / 256); };

  // ---- stage 1: fp32 -> bf16 hi/lo splits ----
  split_x_kernel<<<blocks((long long)BATCH * XCOLS), T, 0, stream>>>(x, Ahi, Alo, Dhi, Dlo);
  split_w_kernel<<<blocks((long long)NGPAD * NGENE), T, 0, stream>>>(dgW, Whi, Wlo, NGCOLS, NGPAD, NGENE);
  split_w_kernel<<<blocks((long long)NDPAD0 * NDRUG), T, 0, stream>>>(dW[0], WDhi, WDlo, NDOUT0, NDPAD0, NDRUG);

  // ---- stage 2: big WMMA GEMMs ----
  // gene: h0[b, t*8+g] = gene . dgW^T + dgb        (123 GFLOP, compute-bound)
  gemm_bf16x3_kernel<<<dim3(NGPAD / 64, BATCH / 128), T, 0, stream>>>(
      Ahi, Alo, Whi, Wlo, h0, dgb, NGENE, NGCOLS, /*tanh=*/0);
  // drug layer 1: zd0 = tanh(drug . dW0^T + db0)   (1.7 GFLOP)
  gemm_bf16x3_kernel<<<dim3(NDPAD0 / 64, BATCH / 128), T, 0, stream>>>(
      Dhi, Dlo, WDhi, WDlo, zd[0], db[0], NDRUG, NDOUT0, /*tanh=*/1);

  // ---- stage 3: tree levels with two-pass BN ----
  term0_kernel<<<blocks((long long)BATCH * Cterm[0]), T, 0, stream>>>(h0, tW[0], tb[0], zt[0]);
  stats_kernel<<<Cterm[0], T, 0, stream>>>(zt[0], mt[0], it[0], Cterm[0]);
  const int nterms[5] = {625, 125, 25, 5, 1};
  for (int i = 1; i < 5; ++i) {
    term_next_kernel<<<blocks((long long)BATCH * Cterm[i]), T, 0, stream>>>(
        zt[i - 1], mt[i - 1], it[i - 1], tg[i - 1], tbe[i - 1],
        tW[i], tb[i], zt[i], nterms[i], Cterm[i - 1]);
    stats_kernel<<<Cterm[i], T, 0, stream>>>(zt[i], mt[i], it[i], Cterm[i]);
  }

  // ---- stage 4: drug branch tail + aux heads ----
  stats_kernel<<<Cdrug[0], T, 0, stream>>>(zd[0], md[0], id[0], Cdrug[0]);
  aux_kernel<<<blocks(BATCH), T, 0, stream>>>(zd[0], md[0], id[0], dg[0], dbe[0],
                                              aW1[0], ab1[0], aW2[0], ab2[0],
                                              out + 1 * BATCH, Cdrug[0]);
  for (int i = 1; i < 3; ++i) {
    fc_bn_tanh_kernel<<<blocks((long long)BATCH * Cdrug[i]), T, 0, stream>>>(
        zd[i - 1], md[i - 1], id[i - 1], dg[i - 1], dbe[i - 1],
        dW[i], db[i], zd[i], Cdrug[i - 1], Cdrug[i]);
    stats_kernel<<<Cdrug[i], T, 0, stream>>>(zd[i], md[i], id[i], Cdrug[i]);
    aux_kernel<<<blocks(BATCH), T, 0, stream>>>(zd[i], md[i], id[i], dg[i], dbe[i],
                                                aW1[i], ab1[i], aW2[i], ab2[i],
                                                out + (i + 1) * BATCH, Cdrug[i]);
  }

  // ---- stage 5: final head ----
  final_z_kernel<<<blocks((long long)BATCH * 6), T, 0, stream>>>(
      zt[4], mt[4], it[4], tg[4], tbe[4],
      zd[2], md[2], id[2], dg[2], dbe[2],
      finW, finb, zf);
  stats_kernel<<<6, T, 0, stream>>>(zf, mf, iff, 6);
  aux_kernel<<<blocks(BATCH), T, 0, stream>>>(zf, mf, iff, fing, finbeta,
                                              finauxW, finauxb, foutW, foutb,
                                              out + 0 * BATCH, 6);
}
